// SigLipAttention_26783416058598
// MI455X (gfx1250) — matrix operand order repaired and checked by reading, not compiled, not run on hardware
//
#include <hip/hip_runtime.h>

typedef __bf16 v16bf __attribute__((ext_vector_type(16)));
typedef __bf16 v8bf  __attribute__((ext_vector_type(8)));
typedef float  v8f   __attribute__((ext_vector_type(8)));
typedef unsigned short u16;

__device__ __forceinline__ u16 f2bf(float f) {
    unsigned u = __float_as_uint(f);
    u += 0x7FFFu + ((u >> 16) & 1u);   // round-to-nearest-even
    return (u16)(u >> 16);
}
__device__ __forceinline__ float bf2f(u16 h) {
    return __uint_as_float((unsigned)h << 16);
}

__device__ __forceinline__ v16bf mkfrag(const u16* lo, const u16* hi) {
    v8bf a = *reinterpret_cast<const v8bf*>(lo);
    v8bf b = *reinterpret_cast<const v8bf*>(hi);
    v16bf r;
#pragma unroll
    for (int i = 0; i < 8; ++i) { r[i] = a[i]; r[i + 8] = b[i]; }
    return r;
}

// ---------------------------------------------------------------------------
// Projection GEMM: C = A(bf16, MxK row-major) * W(bf16, NxK row-major)^T + bias
// M,N,K multiples of 64/64/32; all leading dims = 768.
// ---------------------------------------------------------------------------
constexpr int BM = 64, BN = 64, BK = 32, LDSS = 40;
constexpr int EMBC = 768;

template<bool OUT_BF16>
__global__ __launch_bounds__(128) void gemm_proj(
    const u16* __restrict__ Ap, const u16* __restrict__ Bp,
    const float* __restrict__ bias, void* __restrict__ Cp, int K)
{
    __shared__ u16 sA[BM * LDSS];
    __shared__ u16 sB[BN * LDSS];

    const int tid = threadIdx.x;
    const int m0 = blockIdx.y * BM;
    const int n0 = blockIdx.x * BN;
    const int w = tid >> 5, lane = tid & 31;
    const int wm = (w >> 1) * 32, wn = (w & 1) * 32;

    v8f acc[2][2] = {};

    const int arow = tid >> 1;          // 0..63
    const int acb  = (tid & 1) * 16;    // 0 or 16

    for (int k0 = 0; k0 < K; k0 += BK) {
        {
            const u16* src = Ap + (long long)(m0 + arow) * EMBC + k0 + acb;
            uint4 d0 = reinterpret_cast<const uint4*>(src)[0];
            uint4 d1 = reinterpret_cast<const uint4*>(src)[1];
            reinterpret_cast<uint4*>(&sA[arow * LDSS + acb])[0] = d0;
            reinterpret_cast<uint4*>(&sA[arow * LDSS + acb])[1] = d1;
        }
        {
            const u16* src = Bp + (long long)(n0 + arow) * EMBC + k0 + acb;
            uint4 d0 = reinterpret_cast<const uint4*>(src)[0];
            uint4 d1 = reinterpret_cast<const uint4*>(src)[1];
            reinterpret_cast<uint4*>(&sB[arow * LDSS + acb])[0] = d0;
            reinterpret_cast<uint4*>(&sB[arow * LDSS + acb])[1] = d1;
        }
        __syncthreads();

        const int rl = lane & 15, kh = lane >> 4;
        v16bf af[2], bfr[2];
#pragma unroll
        for (int i = 0; i < 2; ++i) {
            const u16* base = &sA[(wm + i * 16 + rl) * LDSS + kh * 8];
            af[i] = mkfrag(base, base + 16);
        }
#pragma unroll
        for (int j = 0; j < 2; ++j) {
            const u16* base = &sB[(wn + j * 16 + rl) * LDSS + kh * 8];
            bfr[j] = mkfrag(base, base + 16);
        }
#pragma unroll
        for (int i = 0; i < 2; ++i)
#pragma unroll
            for (int j = 0; j < 2; ++j)
                acc[i][j] = __builtin_amdgcn_wmma_f32_16x16x32_bf16(
                    false, af[i], false, bfr[j], (short)0, acc[i][j], false, false);

        __syncthreads();
    }

    const int ccol = lane & 15;
    const int rbase = (lane >> 4) * 8;
#pragma unroll
    for (int i = 0; i < 2; ++i)
#pragma unroll
        for (int j = 0; j < 2; ++j) {
            const int gm = m0 + wm + i * 16 + rbase;
            const int gn = n0 + wn + j * 16 + ccol;
            const float bv = bias[gn];
#pragma unroll
            for (int r = 0; r < 8; ++r) {
                float vv = acc[i][j][r] + bv;
                long long idx = (long long)(gm + r) * EMBC + gn;
                if (OUT_BF16) ((u16*)Cp)[idx] = f2bf(vv);
                else          ((float*)Cp)[idx] = vv;
            }
        }
}

// ---------------------------------------------------------------------------
// Fused attention: per (b,h) head and 64-row Q block:
//   scores = Q K^T * 0.125 (WMMA, into LDS slab), row softmax in LDS,
//   write normalized fp32 probs to attn output once (coalesced),
//   ctx = probs @ V via WMMA with A-fragments read straight from the slab,
//   1/rowsum folded into the PV epilogue.
// ---------------------------------------------------------------------------
constexpr int SQ = 576, HH = 12, DD = 64;
constexpr int SSTR = 584;   // 576 + 8 pad: conflict-free row fans
constexpr int QSTR = 72;    // 64 + 8 pad

__global__ __launch_bounds__(128) void fused_attn(
    const u16* __restrict__ Q, const u16* __restrict__ Km,
    const u16* __restrict__ V, float* __restrict__ attnOut,
    u16* __restrict__ ctx)
{
    __shared__ u16 sS[64 * SSTR];   // score slab (bf16)      74752 B
    __shared__ u16 sQ[64 * QSTR];   // Q block                 9216 B
    __shared__ u16 sW[64 * QSTR];   // K-chunk / V-tile stage  9216 B
    __shared__ float sInv[64];      // per-row 1/sum

    const int tid = threadIdx.x;
    const int z = blockIdx.z;
    const int b = z / HH, h = z % HH;
    const int m0 = blockIdx.x * 64;

    const int w = tid >> 5, lane = tid & 31;
    const int wm = (w >> 1) * 32, wn = (w & 1) * 32;
    const int rl = lane & 15, kh = lane >> 4;
    const int ccol = lane & 15;
    const int rbase = (lane >> 4) * 8;

    const long long headOff = (long long)b * SQ * EMBC + h * DD;

    // ---- phase 0: Q block [64 x 64] -> LDS ----
    const int qrow = tid >> 1;
    const int qcb  = (tid & 1) * 32;
    {
        const u16* src = Q + headOff + (long long)(m0 + qrow) * EMBC + qcb;
#pragma unroll
        for (int q = 0; q < 4; ++q)
            reinterpret_cast<uint4*>(&sQ[qrow * QSTR + qcb])[q] =
                reinterpret_cast<const uint4*>(src)[q];
    }

    // ---- phase 1: scores into LDS slab, 9 chunks of 64 keys ----
    for (int nc = 0; nc < 9; ++nc) {
        {
            const u16* src = Km + headOff + (long long)(nc * 64 + qrow) * EMBC + qcb;
#pragma unroll
            for (int q = 0; q < 4; ++q)
                reinterpret_cast<uint4*>(&sW[qrow * QSTR + qcb])[q] =
                    reinterpret_cast<const uint4*>(src)[q];
        }
        __syncthreads();

        v8f acc[2][2] = {};
#pragma unroll
        for (int k0 = 0; k0 < DD; k0 += BK) {
            v16bf af[2], bfr[2];
#pragma unroll
            for (int i = 0; i < 2; ++i) {
                const u16* base = &sQ[(wm + i * 16 + rl) * QSTR + k0 + kh * 8];
                af[i] = mkfrag(base, base + 16);
            }
#pragma unroll
            for (int j = 0; j < 2; ++j) {
                const u16* base = &sW[(wn + j * 16 + rl) * QSTR + k0 + kh * 8];
                bfr[j] = mkfrag(base, base + 16);
            }
#pragma unroll
            for (int i = 0; i < 2; ++i)
#pragma unroll
                for (int j = 0; j < 2; ++j)
                    acc[i][j] = __builtin_amdgcn_wmma_f32_16x16x32_bf16(
                        false, af[i], false, bfr[j], (short)0, acc[i][j], false, false);
        }
        // spill scaled scores to slab
#pragma unroll
        for (int i = 0; i < 2; ++i)
#pragma unroll
            for (int j = 0; j < 2; ++j)
#pragma unroll
                for (int r = 0; r < 8; ++r)
                    sS[(wm + i * 16 + rbase + r) * SSTR + nc * 64 + wn + j * 16 + ccol] =
                        f2bf(acc[i][j][r] * 0.125f);
        __syncthreads();
    }

    // ---- phase 2: row softmax in LDS (2 threads per row, 288 elems each) ----
    {
        const int row = tid >> 1;
        const int half = tid & 1;
        u16* rp = &sS[row * SSTR + half * 288];
        float m = -3.4e38f;
        for (int i = 0; i < 288; ++i) m = fmaxf(m, bf2f(rp[i]));
        m = fmaxf(m, __shfl_xor(m, 1, 32));
        float s = 0.f;
        for (int i = 0; i < 288; ++i) {
            float e = __expf(bf2f(rp[i]) - m);
            s += e;
            rp[i] = f2bf(e);        // keep unnormalized exp in slab
        }
        s += __shfl_xor(s, 1, 32);
        if (half == 0) sInv[row] = 1.0f / s;
    }
    __syncthreads();

    // ---- phase 2b: coalesced normalized fp32 attn write (once) ----
    {
        float* aout = attnOut + (long long)z * SQ * SQ + (long long)m0 * SQ;
        for (int r = 0; r < 64; ++r) {
            const float iv = sInv[r];
            for (int c = tid; c < SQ; c += 128)
                aout[(long long)r * SQ + c] = bf2f(sS[r * SSTR + c]) * iv;
        }
    }

    // ---- phase 3: ctx = exp-slab @ V, normalization in epilogue ----
    v8f accp[2][2] = {};
    const int kk = tid >> 2;          // 0..31
    const int nb = (tid & 3) * 16;    // 0,16,32,48
    for (int k0 = 0; k0 < SQ; k0 += BK) {
        {   // stage V tile [64 d x 32 k] transposed into sW[n][k]
            const u16* src = V + headOff + (long long)(k0 + kk) * EMBC + nb;
            union { uint4 v; u16 s[8]; } u0, u1;
            u0.v = reinterpret_cast<const uint4*>(src)[0];
            u1.v = reinterpret_cast<const uint4*>(src)[1];
#pragma unroll
            for (int i = 0; i < 8; ++i) {
                sW[(nb + i) * QSTR + kk]     = u0.s[i];
                sW[(nb + 8 + i) * QSTR + kk] = u1.s[i];
            }
        }
        __syncthreads();

        v16bf af[2], bfr[2];
#pragma unroll
        for (int i = 0; i < 2; ++i) {
            const u16* base = &sS[(wm + i * 16 + rl) * SSTR + k0 + kh * 8];
            af[i] = mkfrag(base, base + 16);
        }
#pragma unroll
        for (int j = 0; j < 2; ++j) {
            const u16* base = &sW[(wn + j * 16 + rl) * QSTR + kh * 8];
            bfr[j] = mkfrag(base, base + 16);
        }
#pragma unroll
        for (int i = 0; i < 2; ++i)
#pragma unroll
            for (int j = 0; j < 2; ++j)
                accp[i][j] = __builtin_amdgcn_wmma_f32_16x16x32_bf16(
                    false, af[i], false, bfr[j], (short)0, accp[i][j], false, false);
        __syncthreads();
    }

    // epilogue: scale by 1/rowsum, store bf16 ctx [b, s, h*64+d]
#pragma unroll
    for (int i = 0; i < 2; ++i)
#pragma unroll
        for (int j = 0; j < 2; ++j) {
            const int lm = wm + i * 16 + rbase;
            const int gn = wn + j * 16 + ccol;   // d index 0..63
#pragma unroll
            for (int r = 0; r < 8; ++r) {
                const float vv = accp[i][j][r] * sInv[lm + r];
                ctx[headOff + (long long)(m0 + lm + r) * EMBC + gn] = f2bf(vv);
            }
        }
}

__global__ __launch_bounds__(256) void cvt_f32_to_bf16(const float* __restrict__ s,
                                                       u16* __restrict__ d, long long n) {
    long long i = ((long long)blockIdx.x * 256 + threadIdx.x) * 4;
    if (i >= n) return;
    float4 f = *reinterpret_cast<const float4*>(s + i);
    uint2 p;
    p.x = f2bf(f.x) | ((unsigned)f2bf(f.y) << 16);
    p.y = f2bf(f.z) | ((unsigned)f2bf(f.w) << 16);
    *reinterpret_cast<uint2*>(d + i) = p;
}

extern "C" void kernel_launch(void* const* d_in, const int* in_sizes, int n_in,
                              void* d_out, int out_size, void* d_ws, size_t ws_size,
                              hipStream_t stream) {
    (void)in_sizes; (void)n_in; (void)out_size; (void)ws_size;
    const int B = 32, S = 576, EMB = 768, H = 12;
    const long long M = (long long)B * S;  // 18432

    const float* x  = (const float*)d_in[0];
    const float* qw = (const float*)d_in[1];
    const float* qb = (const float*)d_in[2];
    const float* kw = (const float*)d_in[3];
    const float* kb = (const float*)d_in[4];
    const float* vw = (const float*)d_in[5];
    const float* vb = (const float*)d_in[6];
    const float* ow = (const float*)d_in[7];
    const float* ob = (const float*)d_in[8];

    u16* xb  = (u16*)d_ws;
    u16* wqb = xb  + (size_t)M * EMB;
    u16* wkb = wqb + (size_t)EMB * EMB;
    u16* wvb = wkb + (size_t)EMB * EMB;
    u16* wob = wvb + (size_t)EMB * EMB;
    u16* Qb  = wob + (size_t)EMB * EMB;
    u16* Kb  = Qb  + (size_t)M * EMB;
    u16* Vb  = Kb  + (size_t)M * EMB;
    u16* Cb  = Vb  + (size_t)M * EMB;

    float* outp = (float*)d_out;               // [B,S,EMB]
    float* attn = outp + (size_t)M * EMB;      // [B,H,S,S]

    // stage 0: fp32 -> bf16
    {
        long long n = M * EMB;
        cvt_f32_to_bf16<<<(unsigned)((n / 4 + 255) / 256), 256, 0, stream>>>(x, xb, n);
        long long nw = (long long)EMB * EMB;
        unsigned gw = (unsigned)((nw / 4 + 255) / 256);
        cvt_f32_to_bf16<<<gw, 256, 0, stream>>>(qw, wqb, nw);
        cvt_f32_to_bf16<<<gw, 256, 0, stream>>>(kw, wkb, nw);
        cvt_f32_to_bf16<<<gw, 256, 0, stream>>>(vw, wvb, nw);
        cvt_f32_to_bf16<<<gw, 256, 0, stream>>>(ow, wob, nw);
    }

    dim3 blk(128, 1, 1);
    dim3 gp(EMB / BN, (unsigned)(M / BM), 1);

    // stage 1: QKV projections
    gemm_proj<true><<<gp, blk, 0, stream>>>(xb, wqb, qb, Qb, EMB);
    gemm_proj<true><<<gp, blk, 0, stream>>>(xb, wkb, kb, Kb, EMB);
    gemm_proj<true><<<gp, blk, 0, stream>>>(xb, wvb, vb, Vb, EMB);

    // stage 2: fused scores + softmax + attn-write + PV
    dim3 gf(S / 64, 1, (unsigned)(B * H));  // (9,1,384)
    fused_attn<<<gf, blk, 0, stream>>>(Qb, Kb, Vb, attn, Cb);

    // stage 3: out = ctx @ o_w^T + o_b -> fp32
    gemm_proj<false><<<gp, blk, 0, stream>>>(Cb, wob, ob, outp, EMB);
}
